// Encoder_24850680775135
// MI455X (gfx1250) — compile-verified, matrix operand order checked by reading, not run on hardware
//
#include <hip/hip_runtime.h>
#include <hip/hip_bf16.h>
#include <math.h>

typedef __attribute__((ext_vector_type(16))) _Float16 v16h;
typedef __attribute__((ext_vector_type(8)))  _Float16 v8h;
typedef __attribute__((ext_vector_type(8)))  float    v8f;

#define WPB 8  // waves per block (8 * 32 = 256 threads)

// ---------------- elementwise f32 -> f16 ----------------
__global__ void k_cvt_f16(const float* __restrict__ in, _Float16* __restrict__ out, long n) {
    long i = (long)blockIdx.x * blockDim.x + threadIdx.x;
    if (i < n) out[i] = (_Float16)in[i];
}

// ---------------- weight transpose + convert: in[K,N] f32 -> out[N,K] f16 ----------------
__global__ void k_wt_cvt(const float* __restrict__ in, _Float16* __restrict__ out, int K, int N) {
    long i = (long)blockIdx.x * blockDim.x + threadIdx.x;
    if (i < (long)K * N) {
        int n = (int)(i / K);
        int k = (int)(i % K);
        out[i] = (_Float16)in[(long)k * N + n];
    }
}

// ---------------- V transpose per head-batch: Vh[16][2048][64] -> Vt[16][64][2048] ----------------
__global__ void k_vtrans(const _Float16* __restrict__ vh, _Float16* __restrict__ vt) {
    long i = (long)blockIdx.x * blockDim.x + threadIdx.x;  // 16*64*2048 = 2^21
    if (i < (1L << 21)) {
        int bh  = (int)(i >> 17);
        int rem = (int)(i & 131071);
        int dh  = rem >> 11;
        int s   = rem & 2047;
        vt[i] = vh[((long)bh * 2048 + s) * 64 + dh];
    }
}

// ---- fragment helpers (ISA 16-bit 16x32 A-layout: lane m=lane&15, kh=lane>>4,
//      elements 0..7 -> K = 8*kh + 0..7, elements 8..15 -> K = 16 + 8*kh + 0..7)
__device__ inline v16h frag_f16(const _Float16* p) {
    v8h lo = *(const v8h*)p;
    v8h hi = *(const v8h*)(p + 16);
    v16h a;
#pragma unroll
    for (int i = 0; i < 8; ++i) { a[i] = lo[i]; a[8 + i] = hi[i]; }
    return a;
}
__device__ inline v16h frag_f32(const float* p) {
    v16h a;
#pragma unroll
    for (int i = 0; i < 8; ++i) { a[i] = (_Float16)p[i]; a[8 + i] = (_Float16)p[16 + i]; }
    return a;
}

// ---------------- generic WMMA GEMM (f16 x f16 -> f32) ----------------
// C[M,N] = scale * (A[M,K] @ Bt[N,K]^T) + bias, optional relu. One wave = 16x64 strip.
__global__ __launch_bounds__(256) void k_wmma_gemm(
    const _Float16* __restrict__ A, const _Float16* __restrict__ Bt,
    const float* __restrict__ bias,
    float* __restrict__ C32, _Float16* __restrict__ C16,
    int M, int N, int K, int lda, int ldb, int ldc,
    float scale, int relu)
{
    const int lane = threadIdx.x;
    const int nStrips = N >> 6;
    long gw = (long)blockIdx.x * WPB + threadIdx.y;
    if (gw >= (long)(M >> 4) * nStrips) return;   // wave-uniform exit
    const int m0 = (int)(gw / nStrips) << 4;
    const int n0 = (int)(gw % nStrips) << 6;
    const int m  = lane & 15;
    const int kh = lane >> 4;

    v8f acc[4];
#pragma unroll
    for (int j = 0; j < 4; ++j)
#pragma unroll
        for (int e = 0; e < 8; ++e) acc[j][e] = 0.0f;

    const long aRow = (long)(m0 + m) * lda;
    for (int k0 = 0; k0 < K; k0 += 32) {
        const int koff = k0 + 8 * kh;
        v16h a = frag_f16(A + aRow + koff);
#pragma unroll
        for (int j = 0; j < 4; ++j) {
            v16h b = frag_f16(Bt + (long)(n0 + j * 16 + m) * ldb + koff);
            acc[j] = __builtin_amdgcn_wmma_f32_16x16x32_f16(
                false, a, false, b, (short)0, acc[j], false, false);
        }
    }

#pragma unroll
    for (int j = 0; j < 4; ++j) {
        const int col = n0 + j * 16 + m;
        const float bv = bias ? bias[col] : 0.0f;
#pragma unroll
        for (int r = 0; r < 8; ++r) {
            const int row = m0 + r + 8 * kh;
            float v = acc[j][r] * scale + bv;
            if (relu) v = fmaxf(v, 0.0f);
            const long idx = (long)row * ldc + col;
            if (C32) C32[idx] = v;
            if (C16) C16[idx] = (_Float16)v;
        }
    }
}

// ---------------- fused attention: scores -> softmax -> attn write -> attn@V ----------------
// Block = 8 waves, owns 16 query rows of one head-batch. LDS holds the full 16x2048
// score row-block (128 KB), so attn touches HBM exactly once (the mandatory output write).
// grid = (L/16, 1, B*H)
__global__ __launch_bounds__(256) void k_attn_fused(
    const _Float16* __restrict__ qh, const _Float16* __restrict__ kh_,
    const _Float16* __restrict__ vt,
    float* __restrict__ attn, _Float16* __restrict__ ctx16)
{
    extern __shared__ float smem[];
    float* sc   = smem;                 // [16][2048] scores/attn      (131072 B)
    float* part = smem + 16 * 2048;     // [8][16][64] ctx partials    ( 32768 B)
    float* red  = part + 8 * 16 * 64;   // [16][17] softmax scratch    (  1088 B)

    const int lane = threadIdx.x;
    const int wave = threadIdx.y;
    const int tid  = wave * 32 + lane;
    const int m    = lane & 15;
    const int khs  = lane >> 4;
    const int mT   = blockIdx.x;        // query tile
    const int bh   = blockIdx.z;        // head-batch

    const long qkS  = 131072;           // 2048*64 per head-batch
    const _Float16* Q = qh  + (long)bh * qkS + (long)(mT * 16 + m) * 64;
    const _Float16* Kb = kh_ + (long)bh * qkS;
    const _Float16* Vb = vt  + (long)bh * qkS;   // Vt: [64][2048] per batch

    // ---- phase 1: scores (16 x 2048) via WMMA, scaled, into LDS ----
    const int nBase = wave * 256;
#pragma unroll 4
    for (int j = 0; j < 16; ++j) {
        v8f acc;
#pragma unroll
        for (int e = 0; e < 8; ++e) acc[e] = 0.0f;
        const _Float16* Krow = Kb + (long)(nBase + j * 16 + m) * 64;
#pragma unroll
        for (int k0 = 0; k0 < 64; k0 += 32) {
            v16h a = frag_f16(Q + k0 + 8 * khs);
            v16h b = frag_f16(Krow + k0 + 8 * khs);
            acc = __builtin_amdgcn_wmma_f32_16x16x32_f16(
                false, a, false, b, (short)0, acc, false, false);
        }
#pragma unroll
        for (int r = 0; r < 8; ++r)
            sc[(r + 8 * khs) * 2048 + nBase + j * 16 + m] = acc[r] * 0.125f;
    }
    __syncthreads();

    // ---- phase 2: softmax over each of the 16 rows (in LDS), write attn to d_out ----
    {
        const int row = tid >> 4;       // 0..15
        const int t   = tid & 15;       // 16 threads per row
        float* sr = sc + row * 2048;
        float mx = -INFINITY;
        for (int i = 0; i < 128; ++i) mx = fmaxf(mx, sr[t + i * 16]);
        red[row * 17 + t] = mx;
        __syncthreads();
        if (t == 0) {
            float m2 = red[row * 17];
            for (int u = 1; u < 16; ++u) m2 = fmaxf(m2, red[row * 17 + u]);
            red[row * 17 + 16] = m2;
        }
        __syncthreads();
        mx = red[row * 17 + 16];
        float sum = 0.0f;
        for (int i = 0; i < 128; ++i) {
            const int c = t + i * 16;
            float e = __expf(sr[c] - mx);
            sr[c] = e;
            sum += e;
        }
        __syncthreads();
        red[row * 17 + t] = sum;
        __syncthreads();
        if (t == 0) {
            float s2 = red[row * 17];
            for (int u = 1; u < 16; ++u) s2 += red[row * 17 + u];
            red[row * 17 + 16] = s2;
        }
        __syncthreads();
        const float inv = 1.0f / red[row * 17 + 16];
        float* ga = attn + (long)bh * 4194304 + (long)(mT * 16 + row) * 2048;
        for (int i = 0; i < 128; ++i) {
            const int c = t + i * 16;
            float v = sr[c] * inv;
            sr[c] = v;
            ga[c] = v;          // the one and only HBM pass over attn
        }
    }
    __syncthreads();

    // ---- phase 3: ctx = attn @ V, K=2048 split across the 8 waves ----
    v8f cacc[4];
#pragma unroll
    for (int j = 0; j < 4; ++j)
#pragma unroll
        for (int e = 0; e < 8; ++e) cacc[j][e] = 0.0f;

    for (int k0 = wave * 256; k0 < wave * 256 + 256; k0 += 32) {
        v16h a = frag_f32(sc + m * 2048 + k0 + 8 * khs);   // LDS f32 -> f16 fragment
#pragma unroll
        for (int j = 0; j < 4; ++j) {
            v16h b = frag_f16(Vb + (long)(j * 16 + m) * 2048 + k0 + 8 * khs);
            cacc[j] = __builtin_amdgcn_wmma_f32_16x16x32_f16(
                false, a, false, b, (short)0, cacc[j], false, false);
        }
    }
#pragma unroll
    for (int j = 0; j < 4; ++j)
#pragma unroll
        for (int r = 0; r < 8; ++r)
            part[wave * 1024 + (r + 8 * khs) * 64 + j * 16 + m] = cacc[j][r];
    __syncthreads();

    // ---- cross-wave K reduction, write ctx f16 ----
    for (int o = tid; o < 1024; o += 256) {
        float s = 0.0f;
#pragma unroll
        for (int w = 0; w < 8; ++w) s += part[w * 1024 + o];
        const long idx = (long)bh * qkS + (long)(mT * 16 + (o >> 6)) * 64 + (o & 63);
        ctx16[idx] = (_Float16)s;
    }
}

// ---------------- fused residual + LayerNorm (D = 512, 256 threads) ----------------
__global__ __launch_bounds__(256) void k_layernorm(
    const float* __restrict__ a, const float* __restrict__ b,
    const float* __restrict__ g, const float* __restrict__ be,
    float* __restrict__ out32, _Float16* __restrict__ out16, int D)
{
    const long base = (long)blockIdx.x * D;
    const int tid = threadIdx.x;
    __shared__ float rs[256], rs2[256];

    float vals[2];
    float s = 0.0f, s2 = 0.0f;
#pragma unroll
    for (int i = 0; i < 2; ++i) {
        const int c = tid + 256 * i;
        float v = a[base + c] + b[base + c];
        vals[i] = v;
        s += v; s2 += v * v;
    }
    rs[tid] = s; rs2[tid] = s2; __syncthreads();
    for (int st = 128; st > 0; st >>= 1) {
        if (tid < st) { rs[tid] += rs[tid + st]; rs2[tid] += rs2[tid + st]; }
        __syncthreads();
    }
    const float mean = rs[0] / (float)D;
    const float var  = rs2[0] / (float)D - mean * mean;
    const float rstd = rsqrtf(var + 1e-5f);
#pragma unroll
    for (int i = 0; i < 2; ++i) {
        const int c = tid + 256 * i;
        float y = (vals[i] - mean) * rstd * g[c] + be[c];
        out32[base + c] = y;
        if (out16) out16[base + c] = (_Float16)y;
    }
}

// ---------------- host orchestration ----------------
extern "C" void kernel_launch(void* const* d_in, const int* in_sizes, int n_in,
                              void* d_out, int out_size, void* d_ws, size_t ws_size,
                              hipStream_t stream) {
    (void)in_sizes; (void)n_in; (void)out_size; (void)ws_size;

    const int  D = 512, FF = 2048, DH = 64, BH = 16, Lh = 2048;
    const long BL = 4096;                    // B * L
    const long ED = BL * D;                  // enc element count

    const float* x   = (const float*)d_in[0];
    const float* Wq  = (const float*)d_in[1];  const float* bq  = (const float*)d_in[2];
    const float* Wk  = (const float*)d_in[3];  const float* bk  = (const float*)d_in[4];
    const float* Wv  = (const float*)d_in[5];  const float* bv  = (const float*)d_in[6];
    const float* Wo  = (const float*)d_in[7];  const float* bo  = (const float*)d_in[8];
    const float* g1  = (const float*)d_in[9];  const float* be1 = (const float*)d_in[10];
    const float* W1  = (const float*)d_in[11]; const float* bf1 = (const float*)d_in[12];
    const float* W2  = (const float*)d_in[13]; const float* bf2 = (const float*)d_in[14];
    const float* g2  = (const float*)d_in[15]; const float* be2 = (const float*)d_in[16];

    float* enc  = (float*)d_out;
    float* attn = (float*)d_out + ED;        // [16,2048,2048]

    char* wsb = (char*)d_ws; size_t off = 0;
    auto alloc = [&](size_t bytes) -> void* {
        void* p = wsb + off;
        off = (off + bytes + 255) & ~(size_t)255;
        return p;
    };
    _Float16* xh   = (_Float16*)alloc(BL * D * 2);
    _Float16* wqt  = (_Float16*)alloc((long)D * D * 2);
    _Float16* wkt  = (_Float16*)alloc((long)D * D * 2);
    _Float16* wvt  = (_Float16*)alloc((long)D * D * 2);
    _Float16* wot  = (_Float16*)alloc((long)D * D * 2);
    _Float16* w1t  = (_Float16*)alloc((long)FF * D * 2);
    _Float16* w2t  = (_Float16*)alloc((long)D * FF * 2);
    _Float16* qh   = (_Float16*)alloc(BL * D * 2);
    _Float16* kh_  = (_Float16*)alloc(BL * D * 2);
    _Float16* vh   = (_Float16*)alloc(BL * D * 2);
    _Float16* vt   = (_Float16*)alloc((long)BH * DH * Lh * 2);
    _Float16* ctxh = (_Float16*)alloc(BL * D * 2);
    float*    o32  = (float*)   alloc(BL * D * 4);
    float*    ln1  = (float*)   alloc(BL * D * 4);
    _Float16* ln1h = (_Float16*)alloc(BL * D * 2);
    _Float16* ffh  = (_Float16*)alloc(BL * FF * 2);
    float*    ff2  = (float*)   alloc(BL * D * 4);

    const dim3 gb(32, WPB);
    auto gemmGrid = [](int M, int N) {
        long waves = (long)(M >> 4) * (N >> 6);
        return dim3((unsigned)((waves + WPB - 1) / WPB), 1, 1);
    };

    // 1) precision conversion / weight transposes
    k_cvt_f16<<<(unsigned)((BL * D + 255) / 256), 256, 0, stream>>>(x, xh, BL * D);
    k_wt_cvt<<<(unsigned)(((long)D * D + 255) / 256), 256, 0, stream>>>(Wq, wqt, D, D);
    k_wt_cvt<<<(unsigned)(((long)D * D + 255) / 256), 256, 0, stream>>>(Wk, wkt, D, D);
    k_wt_cvt<<<(unsigned)(((long)D * D + 255) / 256), 256, 0, stream>>>(Wv, wvt, D, D);
    k_wt_cvt<<<(unsigned)(((long)D * D + 255) / 256), 256, 0, stream>>>(Wo, wot, D, D);
    k_wt_cvt<<<(unsigned)(((long)D * FF + 255) / 256), 256, 0, stream>>>(W1, w1t, D, FF);
    k_wt_cvt<<<(unsigned)(((long)FF * D + 255) / 256), 256, 0, stream>>>(W2, w2t, FF, D);

    // 2) Q/K/V projections (f16 out)
    k_wmma_gemm<<<gemmGrid(4096, 512), gb, 0, stream>>>(
        xh, wqt, bq, nullptr, qh, 4096, 512, 512, 512, 512, 512, 1.0f, 0);
    k_wmma_gemm<<<gemmGrid(4096, 512), gb, 0, stream>>>(
        xh, wkt, bk, nullptr, kh_, 4096, 512, 512, 512, 512, 512, 1.0f, 0);
    k_wmma_gemm<<<gemmGrid(4096, 512), gb, 0, stream>>>(
        xh, wvt, bv, nullptr, vh, 4096, 512, 512, 512, 512, 512, 1.0f, 0);

    // 3) V transpose per head-batch
    k_vtrans<<<(unsigned)(((1L << 21) + 255) / 256), 256, 0, stream>>>(vh, vt);

    // 4) fused scores -> softmax -> attn(d_out) -> ctx (one HBM pass over attn)
    {
        const size_t shBytes = (size_t)(16 * 2048 + 8 * 16 * 64 + 16 * 17) * 4;
        k_attn_fused<<<dim3(128, 1, 16), gb, shBytes, stream>>>(qh, kh_, vt, attn, ctxh);
    }

    // 5) out = ctx @ Wo + bo
    k_wmma_gemm<<<gemmGrid(4096, 512), gb, 0, stream>>>(
        ctxh, wot, bo, o32, nullptr, 4096, 512, 512, 512, 512, 512, 1.0f, 0);

    // 6) ln1 = LN(out + x)
    k_layernorm<<<(unsigned)BL, 256, 0, stream>>>(o32, x, g1, be1, ln1, ln1h, D);

    // 7) FFN
    k_wmma_gemm<<<gemmGrid(4096, 2048), gb, 0, stream>>>(
        ln1h, w1t, bf1, nullptr, ffh, 4096, 2048, 512, 512, 512, 2048, 1.0f, 1);
    k_wmma_gemm<<<gemmGrid(4096, 512), gb, 0, stream>>>(
        ffh, w2t, bf2, ff2, nullptr, 4096, 512, 2048, 2048, 2048, 512, 1.0f, 0);

    // 8) enc = LN(ff + ln1) -> d_out
    k_layernorm<<<(unsigned)BL, 256, 0, stream>>>(ff2, ln1, g2, be2, enc, nullptr, D);
}
